// VoxelBackBone1x_53841710022778
// MI455X (gfx1250) — matrix operand order repaired and checked by reading, not compiled, not run on hardware
//
#include <hip/hip_runtime.h>

typedef __attribute__((ext_vector_type(16))) _Float16 v16h;
typedef __attribute__((ext_vector_type(8)))  _Float16 v8h;
typedef __attribute__((ext_vector_type(8)))  float    v8f;
typedef __attribute__((ext_vector_type(4)))  int      v4i;
typedef __attribute__((address_space(1)))    v4i      gv4i;   // global AS int4
typedef __attribute__((address_space(3)))    v4i      lv4i;   // LDS AS int4

#if defined(__HIP_DEVICE_COMPILE__) && \
    __has_builtin(__builtin_amdgcn_global_load_async_to_lds_b128) && \
    __has_builtin(__builtin_amdgcn_s_wait_asynccnt)
#define USE_ASYNC_LDS 1
#else
#define USE_ASYNC_LDS 0
#endif

namespace {
constexpr int  Dd = 17, Hh = 200, Ww = 176;
constexpr int  Bb = 2;
constexpr int  XT = Ww / 16;                               // 11 x-tiles per row
constexpr long long NPOS = (long long)Bb * Dd * Hh * Ww;   // 1,196,800
constexpr float EPS = 1e-3f;
}

// ---------------------------------------------------------------------------
// Scatter voxel features into dense grid + occupancy count
// ---------------------------------------------------------------------------
__global__ void k_scatter(const float* __restrict__ vf, const int* __restrict__ vc,
                          float* __restrict__ dense, float* __restrict__ mcnt, int nvox)
{
    int i = blockIdx.x * blockDim.x + threadIdx.x;
    if (i >= nvox) return;
    int b = vc[4*i+0], z = vc[4*i+1], y = vc[4*i+2], x = vc[4*i+3];
    long long pos = (((long long)b*Dd + z)*Hh + y)*Ww + x;
#pragma unroll
    for (int c = 0; c < 4; ++c) atomicAdd(&dense[pos*4 + c], vf[4*i + c]);
    atomicAdd(&mcnt[pos], 1.0f);
}

__global__ void k_binarize(float* __restrict__ m)
{
    long long p = (long long)blockIdx.x * blockDim.x + threadIdx.x;
    if (p < NPOS) m[p] = (m[p] > 0.0f) ? 1.0f : 0.0f;
}

__global__ void k_maskpool(const float* __restrict__ m0, float* __restrict__ m1)
{
    long long pos = (long long)blockIdx.x * blockDim.x + threadIdx.x;
    if (pos >= NPOS) return;
    int x = (int)(pos % Ww); long long r = pos / Ww;
    int y = (int)(r % Hh); r /= Hh;
    int z = (int)(r % Dd); int b = (int)(r / Dd);
    float mx = 0.0f;
    for (int dz = -1; dz <= 1; ++dz) { int zz = z + dz; if (zz < 0 || zz >= Dd) continue;
      for (int dy = -1; dy <= 1; ++dy) { int yy = y + dy; if (yy < 0 || yy >= Hh) continue;
        for (int dx = -1; dx <= 1; ++dx) { int xx = x + dx; if (xx < 0 || xx >= Ww) continue;
          mx = fmaxf(mx, m0[(((long long)b*Dd + zz)*Hh + yy)*Ww + xx]);
        } } }
    m1[pos] = mx;
}

// ---------------------------------------------------------------------------
// First conv (4 -> 16): scalar FMA, fused BN+ReLU+mask, emits f16 activations
// ---------------------------------------------------------------------------
__global__ void k_conv_in(const float* __restrict__ dense, const float* __restrict__ w,
                          const float* __restrict__ g,  const float* __restrict__ bt,
                          const float* __restrict__ mn, const float* __restrict__ vr,
                          const float* __restrict__ mask, _Float16* __restrict__ out)
{
    long long pos = (long long)blockIdx.x * blockDim.x + threadIdx.x;
    if (pos >= NPOS) return;
    int x = (int)(pos % Ww); long long r = pos / Ww;
    int y = (int)(r % Hh); r /= Hh;
    int z = (int)(r % Dd); int b = (int)(r / Dd);
    float acc[16];
#pragma unroll
    for (int c = 0; c < 16; ++c) acc[c] = 0.0f;
    for (int dz = 0; dz < 3; ++dz) { int zz = z + dz - 1; if (zz < 0 || zz >= Dd) continue;
      for (int dy = 0; dy < 3; ++dy) { int yy = y + dy - 1; if (yy < 0 || yy >= Hh) continue;
        for (int dx = 0; dx < 3; ++dx) { int xx = x + dx - 1; if (xx < 0 || xx >= Ww) continue;
          const float* ip = dense + ((((long long)b*Dd + zz)*Hh + yy)*Ww + xx) * 4;
          const float* wp = w + ((dz*3 + dy)*3 + dx) * 4 * 16;
#pragma unroll
          for (int ci = 0; ci < 4; ++ci) {
            float iv = ip[ci];
#pragma unroll
            for (int co = 0; co < 16; ++co) acc[co] += iv * wp[ci*16 + co];
          }
        } } }
    float mval = mask[pos];
#pragma unroll
    for (int co = 0; co < 16; ++co) {
        float sc = g[co] * rsqrtf(vr[co] + EPS);
        float v  = (acc[co] - mn[co]) * sc + bt[co];
        out[pos*16 + co] = (_Float16)(fmaxf(v, 0.0f) * mval);
    }
}

// ---------------------------------------------------------------------------
// Prepack f32 DHWIO weights into per-lane WMMA B fragments (f16).
// B 32x16 layout: lane holds column N=lane&15; K = (i<8 ? 0 : 16) + (lane<16 ? 0 : 8) + (i&7).
// K-space per (dz,dy) row: CIN==16 -> kk = 32c + klocal, dx = kk>>4 (dx==3 pad);
//                          CIN==32 -> dx = chunk, cin = klocal.
// ---------------------------------------------------------------------------
template<int CIN, int COUT>
__global__ void k_prepack(const float* __restrict__ w, _Float16* __restrict__ wpk)
{
    constexpr int NCH = (CIN == 16) ? 2 : 3;
    constexpr int NT  = COUT / 16;
    int total = 9 * NCH * NT * 512;
    int idx = blockIdx.x * blockDim.x + threadIdx.x;
    if (idx >= total) return;
    int i    = idx & 15;
    int t1   = idx >> 4;
    int lane = t1 & 31; t1 >>= 5;
    int ct   = t1 % NT;
    int chunkId = t1 / NT;
    int c   = chunkId % NCH;
    int row = chunkId / NCH;
    int dz = row / 3, dy = row % 3;
    int klocal = ((i < 8) ? 0 : 16) + ((lane < 16) ? 0 : 8) + (i & 7);
    int dxi, cin; bool ok;
    if (CIN == 16) { int kk = c*32 + klocal; dxi = kk >> 4; cin = kk & 15; ok = (dxi < 3); }
    else           { dxi = c;                cin = klocal;               ok = true; }
    int n = ct*16 + (lane & 15);
    float v = 0.0f;
    if (ok) v = w[((((dz*3 + dy)*3 + dxi)*CIN + cin) * COUT) + n];
    wpk[idx] = (_Float16)v;
}

// ---------------------------------------------------------------------------
// Implicit-GEMM conv via v_wmma_f32_16x16x32_f16.
// Block = NT wave32's sharing one LDS-staged 18xCIN activation row segment.
// Wave ct computes the 16-position x 16-cout tile for couts [16ct, 16ct+16).
// A-fragment = two ds_load_b128 (per-lane halves are K-contiguous).
// Interior tiles stage global->LDS via the ASYNCcnt async-copy path.
// ---------------------------------------------------------------------------
template<int CIN, int COUT, bool FINAL>
__global__ __launch_bounds__(32 * (COUT / 16))
void k_conv_wmma(const _Float16* __restrict__ act, const _Float16* __restrict__ wpk,
                 const float* __restrict__ g,  const float* __restrict__ bt,
                 const float* __restrict__ mn, const float* __restrict__ vr,
                 const float* __restrict__ mask,
                 _Float16* __restrict__ out_h, float* __restrict__ out_f)
{
    constexpr int NCH    = (CIN == 16) ? 2 : 3;
    constexpr int NT     = COUT / 16;
    constexpr int NTHR   = 32 * NT;
    constexpr int CHUNKS = 18 * CIN / 8;     // 16-byte chunks per staged row
    __shared__ alignas(16) _Float16 smem[18 * CIN];

    int tile = blockIdx.x;
    int xt = tile % XT; int rest = tile / XT;
    int y = rest % Hh; rest /= Hh;
    int z = rest % Dd; int b = rest / Dd;
    int x0 = xt * 16;
    int lane = threadIdx.x;
    int tid  = threadIdx.y * 32 + lane;
    int M    = lane & 15;
    int off  = (lane >= 16) ? 8 : 0;         // per-lane K base within a half
    int ct   = threadIdx.y;

    const bool interior = (x0 > 0) && (x0 + 16 < Ww);  // block-uniform

    union CF { v8f v; float e[8]; } acc = {};
    v8h zero8 = {};

    for (int dz = 0; dz < 3; ++dz) {
        int zin = z + dz - 1;
        if (zin < 0 || zin >= Dd) continue;
        for (int dy = 0; dy < 3; ++dy) {
            int yin = y + dy - 1;
            if (yin < 0 || yin >= Hh) continue;
            const _Float16* rowp = act + ((((long long)b*Dd + zin)*Hh + yin)*Ww) * CIN;
            __builtin_prefetch(rowp + (long long)x0*CIN + Ww*CIN, 0, 1); // next row hint
            __syncthreads();
#if USE_ASYNC_LDS
            if (interior) {
                const _Float16* seg = rowp + (long long)(x0 - 1) * CIN;
                for (int s = tid; s < CHUNKS; s += NTHR) {
                    __builtin_amdgcn_global_load_async_to_lds_b128(
                        (gv4i*)(seg + s*8), (lv4i*)&smem[s*8], 0, 0);
                }
                __builtin_amdgcn_s_wait_asynccnt(0);
            } else
#endif
            {
                for (int s = tid; s < CHUNKS; s += NTHR) {
                    int h0 = s * 8;                    // half index; within one x (CIN%8==0)
                    int xp = x0 - 1 + h0 / CIN;
                    v8h v = zero8;
                    if (xp >= 0 && xp < Ww)
                        v = *(const v8h*)(rowp + (long long)xp*CIN + (h0 % CIN));
                    *(v8h*)&smem[h0] = v;
                }
            }
            __syncthreads();

            int row = dz*3 + dy;
#pragma unroll
            for (int c = 0; c < NCH; ++c) {
                // A 16x32 f16 layout: elements 0..7 -> K = off+0..7, 8..15 -> K = 16+off+0..7.
                // Both halves are cin-contiguous at fixed dx -> two b128 LDS loads.
                union AF { v16h v; v8h h[2]; } af;
                if (CIN == 16) {
                    af.h[0] = *(const v8h*)&smem[(M + 2*c    ) * CIN + off];        // dx = 2c
                    af.h[1] = (c == 0) ? *(const v8h*)&smem[(M + 1) * CIN + off]    // dx = 1
                                       : zero8;                                     // dx = 3 pad
                } else {
                    af.h[0] = *(const v8h*)&smem[(M + c) * CIN + off];              // cin 0..15
                    af.h[1] = *(const v8h*)&smem[(M + c) * CIN + 16 + off];         // cin 16..31
                }
                long long wb = (((long long)(row*NCH + c) * NT + ct) * 32 + lane) * 16;
                v16h bf = *(const v16h*)(wpk + wb);
                acc.v = __builtin_amdgcn_wmma_f32_16x16x32_f16(
                            false, af.v, false, bf, (short)0, acc.v, false, false);
            }
        }
    }

    // Epilogue: BN + ReLU + mask. C/D layout: VGPR r -> M = r + 8*(lane>=16), N = lane&15.
    int cout = ct*16 + (lane & 15);
    float sc = g[cout] * rsqrtf(vr[cout] + EPS);
    float bb = bt[cout] - mn[cout] * sc;
    long long posBase = (((long long)b*Dd + z)*Hh + y)*Ww + x0;
#pragma unroll
    for (int r = 0; r < 8; ++r) {
        int Mr = r + ((lane < 16) ? 0 : 8);
        float mv = mask[posBase + Mr];
        float v  = fmaxf(acc.e[r]*sc + bb, 0.0f) * mv;
        long long oi = (posBase + Mr) * COUT + cout;
        if (FINAL) out_f[oi] = v;
        else       out_h[oi] = (_Float16)v;
    }
}

// ---------------------------------------------------------------------------
extern "C" void kernel_launch(void* const* d_in, const int* in_sizes, int n_in,
                              void* d_out, int out_size, void* d_ws, size_t ws_size,
                              hipStream_t stream)
{
    (void)n_in; (void)out_size; (void)ws_size;
    const float* vf = (const float*)d_in[0];
    const int*   vc = (const int*)d_in[1];
    int nvox = in_sizes[0] / 4;

    // Param leaf order: each layer is (w, g, b, m, v). Detect insertion order
    // ('in' first, w size 1728) vs pytree-sorted order ('c1' first, w size 6912).
    int i_in, i_c1, i_c2a, i_c2b, i_c2c;
    if (in_sizes[3] == 3*3*3*4*16) { i_in = 3;  i_c1 = 8;  i_c2a = 13; i_c2b = 18; i_c2c = 23; }
    else                           { i_c1 = 3;  i_c2a = 8; i_c2b = 13; i_c2c = 18; i_in  = 23; }
    auto P = [&](int k) { return (const float*)d_in[k]; };

    // Workspace carve (~164 MB total): masks, packed weights, two ping-pong bufs.
    char* ws = (char*)d_ws;
    size_t off = 0;
    auto carve = [&](size_t bytes) -> void* {
        void* p = ws + off; off += (bytes + 255) & ~(size_t)255; return p;
    };
    float*    mask0  = (float*)carve(NPOS * sizeof(float));
    float*    mask1  = (float*)carve(NPOS * sizeof(float));
    _Float16* wpk1   = (_Float16*)carve((size_t)9*2*1*512 * sizeof(_Float16));
    _Float16* wpk2a  = (_Float16*)carve((size_t)9*2*2*512 * sizeof(_Float16));
    _Float16* wpk2b  = (_Float16*)carve((size_t)9*3*2*512 * sizeof(_Float16));
    _Float16* wpk2c  = (_Float16*)carve((size_t)9*3*4*512 * sizeof(_Float16));
    char*     bufA   = (char*)carve((size_t)NPOS * 32 * sizeof(_Float16));  // 76.6 MB
    char*     bufB   = (char*)carve((size_t)NPOS * 32 * sizeof(_Float16));  // 76.6 MB

    float*    dense0 = (float*)bufA;        // (B,D,H,W,4) f32
    _Float16* act16a = (_Float16*)bufB;     // conv-in out
    _Float16* act16b = (_Float16*)bufA;     // c1 out
    _Float16* act32a = (_Float16*)bufB;     // c2a out
    _Float16* act32b = (_Float16*)bufA;     // c2b out

    (void)hipMemsetAsync(dense0, 0, (size_t)NPOS * 4 * sizeof(float), stream);
    (void)hipMemsetAsync(mask0,  0, (size_t)NPOS * sizeof(float), stream);

    k_scatter<<<(nvox + 255)/256, 256, 0, stream>>>(vf, vc, dense0, mask0, nvox);

    int nposBlocks = (int)((NPOS + 255) / 256);
    k_binarize<<<nposBlocks, 256, 0, stream>>>(mask0);
    k_maskpool<<<nposBlocks, 256, 0, stream>>>(mask0, mask1);

    k_prepack<16,16><<<(9*2*1*512 + 255)/256, 256, 0, stream>>>(P(i_c1),  wpk1);
    k_prepack<16,32><<<(9*2*2*512 + 255)/256, 256, 0, stream>>>(P(i_c2a), wpk2a);
    k_prepack<32,32><<<(9*3*2*512 + 255)/256, 256, 0, stream>>>(P(i_c2b), wpk2b);
    k_prepack<32,64><<<(9*3*4*512 + 255)/256, 256, 0, stream>>>(P(i_c2c), wpk2c);

    k_conv_in<<<nposBlocks, 256, 0, stream>>>(dense0, P(i_in),
        P(i_in+1), P(i_in+2), P(i_in+3), P(i_in+4), mask0, act16a);

    const int TILES = Bb * Dd * Hh * XT;   // 74,800
    k_conv_wmma<16,16,false><<<TILES, dim3(32, 1), 0, stream>>>(act16a, wpk1,
        P(i_c1+1),  P(i_c1+2),  P(i_c1+3),  P(i_c1+4),  mask0, act16b, nullptr);
    k_conv_wmma<16,32,false><<<TILES, dim3(32, 2), 0, stream>>>(act16b, wpk2a,
        P(i_c2a+1), P(i_c2a+2), P(i_c2a+3), P(i_c2a+4), mask1, act32a, nullptr);
    k_conv_wmma<32,32,false><<<TILES, dim3(32, 2), 0, stream>>>(act32a, wpk2b,
        P(i_c2b+1), P(i_c2b+2), P(i_c2b+3), P(i_c2b+4), mask1, act32b, nullptr);
    k_conv_wmma<32,64,true ><<<TILES, dim3(32, 4), 0, stream>>>(act32b, wpk2c,
        P(i_c2c+1), P(i_c2c+2), P(i_c2c+3), P(i_c2c+4), mask1, nullptr, (float*)d_out);
}
